// SpeedrunGPT2Block_59047210385998
// MI455X (gfx1250) — matrix operand order repaired and checked by reading, not compiled, not run on hardware
//
#include <hip/hip_runtime.h>
#include <hip/hip_bf16.h>

// ---------------------------------------------------------------------------
// MI455X (gfx1250) transformer block.
//
// Roofline: ~137 GFLOP of matmul vs ~0.4 GB HBM traffic -> compute bound in
// fp32 VALU, so all GEMMs + both attention matmuls run on
// v_wmma_f32_16x16x32_bf16 (K=32/instr, f32 accumulate, fp32 exponent range).
//
// Data movement (CDNA5-specific):
//  * Weights are converted fp32->bf16 ONCE and stored TRANSPOSED (N-major,
//    Wt[n][k]) so that both GEMM operand tiles (A: 128 rows x 32 k,
//    B: 128 n x 32 k, row stride K) are plain 2D tiles.
//  * Both tiles are staged into LDS by the Tensor Data Mover
//    (tensor_load_to_lds, wave-level, EXEC-ignored), double buffered:
//    wave 0 issues the TDM pair for tile t+1, then s_wait_tensorcnt(2)
//    so tile t's DMA is complete while t+1 overlaps the 8 WMMAs.
//  * Attention: K rows ([b,h,s,hd]) are already in WMMA B-fragment layout
//    (lane n = key, 16 contiguous k per lane); V is written pre-transposed
//    ([b,h,hd,S]) so PV B-fragments are contiguous global loads. Only the
//    16x32 P tile round-trips through LDS (C-layout -> A-layout).
// ---------------------------------------------------------------------------

#define H_   8
#define HD_  128
#define D_   1024
#define S_   2048
#define B_   2
#define BS_  (B_ * S_)
#define EOT_ 50256
#define SCALE_ 0.12f

typedef __attribute__((ext_vector_type(8)))  __bf16 v8bf;
typedef __attribute__((ext_vector_type(16))) __bf16 v16bf;
typedef __attribute__((ext_vector_type(8)))  float  v8f;
typedef __attribute__((ext_vector_type(4)))  unsigned int v4u;
typedef __attribute__((ext_vector_type(8)))  int v8i;
typedef __attribute__((ext_vector_type(4)))  int v4i;

union Frag16 { v16bf v; v8bf h[2]; };

__device__ inline unsigned short f2bf(float f) {
  union { float f; unsigned u; } x; x.f = f;
  unsigned r = x.u + 0x7FFFu + ((x.u >> 16) & 1u);  // round-to-nearest-even
  return (unsigned short)(r >> 16);
}

__device__ inline v8f vzero8() {
  v8f z;
#pragma unroll
  for (int i = 0; i < 8; ++i) z[i] = 0.0f;
  return z;
}

__device__ inline v8f wmma_bf16(v16bf a, v16bf b, v8f c) {
  // D = A(16x32 bf16) * B(32x16 bf16) + C(16x16 f32)
  return __builtin_amdgcn_wmma_f32_16x16x32_bf16(
      /*neg_a=*/false, a, /*neg_b=*/false, b,
      /*c_mod=*/(short)0, c, /*reuse_a=*/false, /*reuse_b=*/false);
}

// ---------------------------------------------------------------------------
// Tensor Data Mover: 2D tile (tile_w x tile_h of 2-byte elems, row stride
// row_stride elems) from global -> LDS, row-major packed in LDS.
// D# built per CDNA5 ISA; groups 2/3 zero
// (<=2D tensor). 6-arg builtin form (clang-23 / therock headers).
// Issued once per wave; tracked by TENSORcnt.
// ---------------------------------------------------------------------------
__device__ inline void tdm_load_2d(unsigned lds_addr, const void* gaddr,
                                   unsigned tile_w, unsigned tile_h,
                                   unsigned row_stride) {
  unsigned long long ga = (unsigned long long)gaddr;
  v4u g0;
  g0[0] = 1u;                                   // count=1 (valid), user mode
  g0[1] = lds_addr;                             // lds_addr        [63:32]
  g0[2] = (unsigned)(ga & 0xFFFFFFFFu);         // global_addr lo  [95:64]
  g0[3] = (unsigned)((ga >> 32) & 0x01FFFFFFu)  // global_addr hi  [120:96]
          | (2u << 30);                         // type=2 ("image")[127:126]
  v8i g1;
  g1[0] = (int)(1u << 16);                      // data_size=1 (2B), mask=0
  g1[1] = (int)((tile_w & 0xFFFFu) << 16);      // tensor_dim0 lo16 @ [63:48]
  g1[2] = (int)(((tile_w >> 16) & 0xFFFFu)      // tensor_dim0 hi16 @ [79:64]
          | ((tile_h & 0xFFFFu) << 16));        // tensor_dim1 lo16 @ [95:80]
  g1[3] = (int)(((tile_h >> 16) & 0xFFFFu)      // tensor_dim1 hi16 @[111:96]
          | ((tile_w & 0xFFFFu) << 16));        // tile_dim0       @[127:112]
  g1[4] = (int)(tile_h & 0xFFFFu);              // tile_dim1; tile_dim2=0
  g1[5] = (int)row_stride;                      // tensor_dim0_stride[31:0]
  g1[6] = 0;                                    // stride hi / dim1_stride
  g1[7] = 0;
  v4i z4 = {0, 0, 0, 0};
  v8i z8;
#pragma unroll
  for (int i = 0; i < 8; ++i) z8[i] = 0;
  __builtin_amdgcn_tensor_load_to_lds(g0, g1, z4, z4, z8, 0);
}

// ---------------------------------------------------------------------------
// fp32 -> bf16 bulk conversion (plain, for value path staging etc.)
// ---------------------------------------------------------------------------
__global__ void f32_to_bf16_k(const float* __restrict__ in,
                              unsigned short* __restrict__ out, int n) {
  int i = blockIdx.x * blockDim.x + threadIdx.x;
  int stride = gridDim.x * blockDim.x;
  for (; i < n; i += stride) out[i] = f2bf(in[i]);
}

// fp32 [K][N] -> bf16 TRANSPOSED [N][K]  (one-time per weight; moves the
// GEMM B transpose out of the hot loop so TDM can stage plain 2D tiles)
__global__ void f32_to_bf16_t_k(const float* __restrict__ in,
                                unsigned short* __restrict__ out, int K,
                                int N) {
  size_t i = (size_t)blockIdx.x * blockDim.x + threadIdx.x;
  size_t total = (size_t)K * N;
  size_t stride = (size_t)gridDim.x * blockDim.x;
  for (; i < total; i += stride) {
    int k = (int)(i / N), n = (int)(i % N);
    out[(size_t)n * K + k] = f2bf(in[i]);
  }
}

// ---------------------------------------------------------------------------
// docs[i] = cumsum(input_ids == EOT)   (tiny sequential scan, S=2048)
// ---------------------------------------------------------------------------
__global__ void docs_k(const int* __restrict__ ids, int* __restrict__ docs) {
  if (threadIdx.x == 0 && blockIdx.x == 0) {
    int c = 0;
    for (int i = 0; i < S_; ++i) {
      c += (ids[i] == EOT_) ? 1 : 0;
      docs[i] = c;
    }
  }
}

// ---------------------------------------------------------------------------
// Residual mix + LayerNorm. One block (256 thr) per row of 1024.
// ---------------------------------------------------------------------------
__global__ __launch_bounds__(256) void mix_ln_k(
    const float* __restrict__ x, const float* __restrict__ x0,
    const float* __restrict__ lambdas, const float* __restrict__ w,
    float* __restrict__ xr, unsigned short* __restrict__ xn) {
  __shared__ float red[256];
  const int row = blockIdx.x;
  const int tid = threadIdx.x;
  const float* xp = x + (size_t)row * D_;
  float l0 = 1.0f, l1 = 0.0f;
  if (lambdas) { l0 = lambdas[0]; l1 = lambdas[1]; }
  float v[4];
  float s = 0.0f;
#pragma unroll
  for (int i = 0; i < 4; ++i) {
    int c = i * 256 + tid;
    float xv = xp[c];
    if (x0) xv = l0 * xv + l1 * x0[(size_t)row * D_ + c];
    if (xr) xr[(size_t)row * D_ + c] = xv;
    v[i] = xv;
    s += xv;
  }
  red[tid] = s;
  __syncthreads();
  for (int off = 128; off > 0; off >>= 1) {
    if (tid < off) red[tid] += red[tid + off];
    __syncthreads();
  }
  float mu = red[0] * (1.0f / D_);
  __syncthreads();
  float s2 = 0.0f;
#pragma unroll
  for (int i = 0; i < 4; ++i) { float d = v[i] - mu; s2 += d * d; }
  red[tid] = s2;
  __syncthreads();
  for (int off = 128; off > 0; off >>= 1) {
    if (tid < off) red[tid] += red[tid + off];
    __syncthreads();
  }
  float inv = rsqrtf(red[0] * (1.0f / D_) + 1e-6f);
#pragma unroll
  for (int i = 0; i < 4; ++i) {
    int c = i * 256 + tid;
    xn[(size_t)row * D_ + c] = f2bf((v[i] - mu) * inv * w[c]);
  }
}

// ---------------------------------------------------------------------------
// bf16 WMMA GEMM with TDM-staged, double-buffered LDS tiles.
// C[M,N] = A[M,K] @ Bt[N,K]^T   (A row-major bf16, Bt N-major bf16).
// Block tile 128x128, 8 waves 4(M)x2(N), each wave 32x64 = 2x4 WMMA tiles.
// Per K-slice of 32: wave 0 issues tensor_load_to_lds for the NEXT A/B
// tiles, then s_wait_tensorcnt(2) (TDM ops of a wave complete in order) so
// the current tiles are ready while the next pair DMAs in the background.
// mode 0: store f32   mode 1: f32 + residual   mode 2: relu^2 -> bf16
// ---------------------------------------------------------------------------
__global__ __launch_bounds__(256) void gemm_bf16_k(
    const unsigned short* __restrict__ A, const unsigned short* __restrict__ Bt,
    const float* __restrict__ resid, float* __restrict__ outF,
    unsigned short* __restrict__ outBF, int M, int N, int K, int mode) {
  __shared__ __align__(16) unsigned short sA[2][128 * 32];
  __shared__ __align__(16) unsigned short sB[2][128 * 32];  // [n][k] tiles
  const int tid = threadIdx.x;
  const int lane = tid & 31;
  const int wid = tid >> 5;
  const int wm = (wid >> 1) * 32;
  const int wn = (wid & 1) * 64;
  const int m0 = blockIdx.y * 128;
  const int n0 = blockIdx.x * 128;

  v8f acc[2][4];
#pragma unroll
  for (int i = 0; i < 2; ++i)
#pragma unroll
    for (int j = 0; j < 4; ++j) acc[i][j] = vzero8();

  const int rA = lane & 15;
  const int kba = (lane >> 4) * 16;   // A-fragment k-base (contiguous)
  const int kbb = (lane >> 4) * 16;   // B-fragment k-base

  const unsigned ldsA0 = (unsigned)(size_t)(void*)sA[0];
  const unsigned ldsA1 = (unsigned)(size_t)(void*)sA[1];
  const unsigned ldsB0 = (unsigned)(size_t)(void*)sB[0];
  const unsigned ldsB1 = (unsigned)(size_t)(void*)sB[1];

  const unsigned short* Abase = A + (size_t)m0 * K;
  const unsigned short* Bbase = Bt + (size_t)n0 * K;

  const int T = K >> 5;  // K-slices of 32
  if (wid == 0) {        // TDM is wave-level (EXEC ignored); wave 0 drives it
    tdm_load_2d(ldsA0, Abase, 32u, 128u, (unsigned)K);
    tdm_load_2d(ldsB0, Bbase, 32u, 128u, (unsigned)K);
  }
  for (int t = 0; t < T; ++t) {
    const int cur = t & 1;
    if (wid == 0) {
      if (t + 1 < T) {  // prefetch next K-slice via TDM, then wait current
        tdm_load_2d(cur ? ldsA0 : ldsA1, Abase + (t + 1) * 32, 32u, 128u,
                    (unsigned)K);
        tdm_load_2d(cur ? ldsB0 : ldsB1, Bbase + (t + 1) * 32, 32u, 128u,
                    (unsigned)K);
        __builtin_amdgcn_s_wait_tensorcnt((short)2);
      } else {
        __builtin_amdgcn_s_wait_tensorcnt((short)0);
      }
    }
    __syncthreads();  // tiles[cur] visible to all waves

    Frag16 af[2];
#pragma unroll
    for (int i = 0; i < 2; ++i) {
      const unsigned short* p = sA[cur] + (wm + i * 16 + rA) * 32;
      af[i].h[0] = *(const v8bf*)(p + kba);
      af[i].h[1] = *(const v8bf*)(p + kba + 8);
    }
#pragma unroll
    for (int j = 0; j < 4; ++j) {
      Frag16 bf;
      bf.v = *(const v16bf*)(sB[cur] + (wn + j * 16 + rA) * 32 + kbb);
#pragma unroll
      for (int i = 0; i < 2; ++i)
        acc[i][j] = wmma_bf16(af[i].v, bf.v, acc[i][j]);
    }
    __syncthreads();  // all readers done before this buffer is re-DMAed
  }

  // C layout: lane holds column n = lane&15, rows m = 8*(lane>>4)+r
  const int cn = lane & 15;
  const int cm = (lane >> 4) * 8;
#pragma unroll
  for (int i = 0; i < 2; ++i)
#pragma unroll
    for (int j = 0; j < 4; ++j)
#pragma unroll
      for (int r = 0; r < 8; ++r) {
        const size_t gm = (size_t)(m0 + wm + i * 16 + cm + r);
        const size_t gn = (size_t)(n0 + wn + j * 16 + cn);
        float v = acc[i][j][r];
        if (mode == 0) {
          outF[gm * N + gn] = v;
        } else if (mode == 1) {
          outF[gm * N + gn] = v + resid[gm * N + gn];
        } else {
          float rl = fmaxf(v, 0.0f);
          outBF[gm * N + gn] = f2bf(rl * rl);
        }
      }
}

// ---------------------------------------------------------------------------
// RoPE (half-truncated) + rmsnorm on q,k; v = sa0*v + sa1*value_embed.
// One block per token, wave w = head w. q,k bf16 [b,h,s,hd]; v bf16
// TRANSPOSED [b,h,hd,s] so attention PV B-fragments are contiguous.
// ---------------------------------------------------------------------------
__global__ __launch_bounds__(256) void rope_rms_v_k(
    const float* __restrict__ qkv, const float* __restrict__ ve,
    const float* __restrict__ sa, unsigned short* __restrict__ qb,
    unsigned short* __restrict__ kb, unsigned short* __restrict__ vt) {
  const int s = blockIdx.x, b = blockIdx.y;
  const int tid = threadIdx.x;
  const int h = tid >> 5, ln = tid & 31;
  const float* base = qkv + (size_t)(b * S_ + s) * (3 * D_) + h * HD_;
  float qo[4], ko[4];
#pragma unroll
  for (int i = 0; i < 2; ++i) {
    int d = ln * 2 + i;  // rotation pair (d, d+64)
    float c = 1.0f, sn = 0.0f;
    if (d < 32) {  // linspace(0,1,32) exponents; upper 32 freqs zero
      float fr = __powf(1024.0f, -(float)d * (1.0f / 31.0f));
      float th = fr * (float)s;
      c = __cosf(th);
      sn = __sinf(th);
    }
    float q1 = base[d], q2 = base[d + 64];
    qo[i] = q1 * c + q2 * sn;
    qo[i + 2] = -q1 * sn + q2 * c;
    float k1 = base[D_ + d], k2 = base[D_ + d + 64];
    ko[i] = k1 * c + k2 * sn;
    ko[i + 2] = -k1 * sn + k2 * c;
  }
  float sq = 0.0f, sk = 0.0f;
#pragma unroll
  for (int i = 0; i < 4; ++i) { sq += qo[i] * qo[i]; sk += ko[i] * ko[i]; }
#pragma unroll
  for (int m = 1; m < 32; m <<= 1) {
    sq += __shfl_xor(sq, m, 32);
    sk += __shfl_xor(sk, m, 32);
  }
  float qs = rsqrtf(sq * (1.0f / HD_) + 1.1920929e-7f);
  float ks = rsqrtf(sk * (1.0f / HD_) + 1.1920929e-7f);
  unsigned short* qrow = qb + ((size_t)(b * H_ + h) * S_ + s) * HD_;
  unsigned short* krow = kb + ((size_t)(b * H_ + h) * S_ + s) * HD_;
#pragma unroll
  for (int i = 0; i < 2; ++i) {
    int d = ln * 2 + i;
    qrow[d] = f2bf(qo[i] * qs);
    qrow[d + 64] = f2bf(qo[i + 2] * qs);
    krow[d] = f2bf(ko[i] * ks);
    krow[d + 64] = f2bf(ko[i + 2] * ks);
  }
  const float sa0 = sa[0], sa1 = sa[1];
#pragma unroll
  for (int j = 0; j < 4; ++j) {
    int d = ln * 4 + j;
    float vv = sa0 * base[2 * D_ + d] + sa1 * ve[(size_t)s * D_ + h * HD_ + d];
    vt[((size_t)(b * H_ + h) * HD_ + d) * S_ + s] = f2bf(vv);
  }
}

// ---------------------------------------------------------------------------
// FlashAttention with causal + document mask. One wave per 16-query tile.
// ---------------------------------------------------------------------------
__global__ __launch_bounds__(32) void attn_k(
    const unsigned short* __restrict__ qg, const unsigned short* __restrict__ kg,
    const unsigned short* __restrict__ vtg, const int* __restrict__ docs,
    unsigned short* __restrict__ out) {
  __shared__ __align__(16) unsigned short P[16 * 32];
  const int qt = blockIdx.x, h = blockIdx.y, b = blockIdx.z;
  const int lane = threadIdx.x;
  const int q0 = qt * 16;
  const unsigned short* qb = qg + (size_t)(b * H_ + h) * S_ * HD_;
  const unsigned short* kb = kg + (size_t)(b * H_ + h) * S_ * HD_;
  const unsigned short* vb = vtg + (size_t)(b * H_ + h) * HD_ * S_;

  const int rA = lane & 15;
  const int kba = (lane >> 4) * 16;
  const int cn = lane & 15;
  const int cm = (lane >> 4) * 8;

  Frag16 qf[4];  // Q 16xHD as 4 A-fragments
#pragma unroll
  for (int c = 0; c < 4; ++c) {
    const unsigned short* p = qb + (size_t)(q0 + rA) * HD_ + c * 32 + kba;
    qf[c].h[0] = *(const v8bf*)p;
    qf[c].h[1] = *(const v8bf*)(p + 8);
  }
  int dq[8];
#pragma unroll
  for (int r = 0; r < 8; ++r) dq[r] = docs[q0 + cm + r];

  float mrow[8], lrow[8];
  v8f oacc[8];
#pragma unroll
  for (int r = 0; r < 8; ++r) { mrow[r] = -3.0e38f; lrow[r] = 0.0f; }
#pragma unroll
  for (int nt = 0; nt < 8; ++nt) oacc[nt] = vzero8();

  for (int j0 = 0; j0 <= q0 + 15; j0 += 32) {  // causal key-tile loop
    v8f sacc[2];
    sacc[0] = vzero8();
    sacc[1] = vzero8();
#pragma unroll
    for (int t = 0; t < 2; ++t) {
      int key = j0 + t * 16 + cn;
      int keyc = key < S_ ? key : S_ - 1;  // clamped reads are masked below
      const unsigned short* krow = kb + (size_t)keyc * HD_;
#pragma unroll
      for (int c = 0; c < 4; ++c) {
        Frag16 bf;
        bf.v = *(const v16bf*)(krow + c * 32 + (lane >> 4) * 16);
        sacc[t] = wmma_bf16(qf[c].v, bf.v, sacc[t]);
      }
    }

    float smax[8], nm[8], alpha[8], rsum[8];
#pragma unroll
    for (int r = 0; r < 8; ++r) smax[r] = -3.0e38f;
#pragma unroll
    for (int t = 0; t < 2; ++t) {
      int key = j0 + t * 16 + cn;
      int dk = docs[key < S_ ? key : S_ - 1];
#pragma unroll
      for (int r = 0; r < 8; ++r) {
        int qi = q0 + cm + r;
        bool ok = (key <= qi) && (dk == dq[r]);
        float sc = ok ? sacc[t][r] * SCALE_ : -3.0e38f;
        sacc[t][r] = sc;
        smax[r] = fmaxf(smax[r], sc);
      }
    }
#pragma unroll
    for (int m = 1; m <= 8; m <<= 1)
#pragma unroll
      for (int r = 0; r < 8; ++r)
        smax[r] = fmaxf(smax[r], __shfl_xor(smax[r], m, 32));
#pragma unroll
    for (int r = 0; r < 8; ++r) {
      float newm = fmaxf(mrow[r], smax[r]);
      float nmf = (newm < -1.0e37f) ? 0.0f : newm;
      alpha[r] = __expf(mrow[r] - nmf);
      mrow[r] = newm;
      nm[r] = nmf;
      rsum[r] = 0.0f;
    }
#pragma unroll
    for (int t = 0; t < 2; ++t)
#pragma unroll
      for (int r = 0; r < 8; ++r) {
        float p = __expf(sacc[t][r] - nm[r]);  // masked -> 0
        rsum[r] += p;
        P[(cm + r) * 32 + t * 16 + cn] = f2bf(p);
      }
#pragma unroll
    for (int m = 1; m <= 8; m <<= 1)
#pragma unroll
      for (int r = 0; r < 8; ++r) rsum[r] += __shfl_xor(rsum[r], m, 32);
#pragma unroll
    for (int r = 0; r < 8; ++r) lrow[r] = lrow[r] * alpha[r] + rsum[r];
#pragma unroll
    for (int nt = 0; nt < 8; ++nt)
#pragma unroll
      for (int r = 0; r < 8; ++r) oacc[nt][r] *= alpha[r];
    __syncthreads();  // LDS transpose C-layout -> A-layout

    Frag16 pf;
    pf.h[0] = *(const v8bf*)(P + rA * 32 + kba);
    pf.h[1] = *(const v8bf*)(P + rA * 32 + kba + 8);
    int kk = j0 + (lane >> 4) * 16;
    if (kk > S_ - 16) kk = S_ - 16;  // OOB keys have p==0
#pragma unroll
    for (int nt = 0; nt < 8; ++nt) {
      Frag16 vf;
      vf.v = *(const v16bf*)(vb + (size_t)(nt * 16 + cn) * S_ + kk);
      oacc[nt] = wmma_bf16(pf.v, vf.v, oacc[nt]);
    }
    __syncthreads();
  }

#pragma unroll
  for (int nt = 0; nt < 8; ++nt)
#pragma unroll
    for (int r = 0; r < 8; ++r) {
      float o = oacc[nt][r] / fmaxf(lrow[r], 1e-20f);
      size_t gm = (size_t)(b * S_ + q0 + cm + r);
      out[gm * D_ + h * HD_ + nt * 16 + cn] = f2bf(o);
    }
}

// ---------------------------------------------------------------------------
extern "C" void kernel_launch(void* const* d_in, const int* in_sizes, int n_in,
                              void* d_out, int out_size, void* d_ws,
                              size_t ws_size, hipStream_t stream) {
  const float* x = (const float*)d_in[0];
  const float* ve = (const float*)d_in[1];
  const float* x0 = (const float*)d_in[2];
  const float* lam = (const float*)d_in[3];
  const float* sal = (const float*)d_in[4];
  const float* ln1 = (const float*)d_in[5];
  const float* ln2 = (const float*)d_in[6];
  const float* wqkv = (const float*)d_in[7];
  const float* wo = (const float*)d_in[8];
  const float* wfc = (const float*)d_in[9];
  const float* wproj = (const float*)d_in[10];
  const int* ids = (const int*)d_in[11];

  char* p = (char*)d_ws;
  auto alloc = [&](size_t bytes) {
    char* r = p;
    p += (bytes + 255) & ~(size_t)255;
    return r;
  };
  unsigned short* wqkv_bt = (unsigned short*)alloc((size_t)3 * D_ * D_ * 2);
  unsigned short* wo_bt = (unsigned short*)alloc((size_t)D_ * D_ * 2);
  unsigned short* wfc_bt = (unsigned short*)alloc((size_t)4 * D_ * D_ * 2);
  unsigned short* wproj_bt = (unsigned short*)alloc((size_t)4 * D_ * D_ * 2);
  float* xr = (float*)alloc((size_t)BS_ * D_ * 4);
  unsigned short* xn_bf = (unsigned short*)alloc((size_t)BS_ * D_ * 2);
  float* qkv = (float*)alloc((size_t)BS_ * 3 * D_ * 4);
  unsigned short* q_bf = (unsigned short*)alloc((size_t)BS_ * D_ * 2);
  unsigned short* k_bf = (unsigned short*)alloc((size_t)BS_ * D_ * 2);
  unsigned short* vt_bf = (unsigned short*)alloc((size_t)BS_ * D_ * 2);
  unsigned short* attn_bf = (unsigned short*)alloc((size_t)BS_ * D_ * 2);
  float* x2 = (float*)alloc((size_t)BS_ * D_ * 4);
  unsigned short* xn2_bf = (unsigned short*)alloc((size_t)BS_ * D_ * 2);
  unsigned short* h_bf = (unsigned short*)alloc((size_t)BS_ * 4 * D_ * 2);
  int* docs = (int*)alloc((size_t)S_ * 4);

  // weights -> bf16, TRANSPOSED to N-major (one time; L2-resident after)
  f32_to_bf16_t_k<<<512, 256, 0, stream>>>(wqkv, wqkv_bt, D_, 3 * D_);
  f32_to_bf16_t_k<<<512, 256, 0, stream>>>(wo, wo_bt, D_, D_);
  f32_to_bf16_t_k<<<512, 256, 0, stream>>>(wfc, wfc_bt, D_, 4 * D_);
  f32_to_bf16_t_k<<<512, 256, 0, stream>>>(wproj, wproj_bt, 4 * D_, D_);
  docs_k<<<1, 32, 0, stream>>>(ids, docs);

  // x = l0*x + l1*x0 ; xn = LN(x)*ln1_w
  mix_ln_k<<<BS_, 256, 0, stream>>>(x, x0, lam, ln1, xr, xn_bf);
  // qkv = xn @ w_qkv
  gemm_bf16_k<<<dim3(3 * D_ / 128, BS_ / 128), 256, 0, stream>>>(
      xn_bf, wqkv_bt, nullptr, qkv, nullptr, BS_, 3 * D_, D_, 0);
  // RoPE + rmsnorm + value-embed mix (v stored transposed)
  rope_rms_v_k<<<dim3(S_, B_), 256, 0, stream>>>(qkv, ve, sal, q_bf, k_bf,
                                                 vt_bf);
  // flash attention
  attn_k<<<dim3(S_ / 16, H_, B_), 32, 0, stream>>>(q_bf, k_bf, vt_bf, docs,
                                                   attn_bf);
  // x2 = xr + attn @ w_o
  gemm_bf16_k<<<dim3(D_ / 128, BS_ / 128), 256, 0, stream>>>(
      attn_bf, wo_bt, xr, x2, nullptr, BS_, D_, D_, 1);
  // xn2 = LN(x2)*ln2_w
  mix_ln_k<<<BS_, 256, 0, stream>>>(x2, nullptr, nullptr, ln2, nullptr,
                                    xn2_bf);
  // h = relu(xn2 @ w_fc)^2
  gemm_bf16_k<<<dim3(4 * D_ / 128, BS_ / 128), 256, 0, stream>>>(
      xn2_bf, wfc_bt, nullptr, nullptr, h_bf, BS_, 4 * D_, D_, 2);
  // out = x2 + h @ w_proj
  gemm_bf16_k<<<dim3(D_ / 128, BS_ / 128), 256, 0, stream>>>(
      h_bf, wproj_bt, x2, (float*)d_out, nullptr, BS_, D_, 4 * D_, 1);
}
